// BondGraphTransformer_88914412961907
// MI455X (gfx1250) — matrix operand order repaired and checked by reading, not compiled, not run on hardware
//
#include <hip/hip_runtime.h>

typedef __attribute__((ext_vector_type(16))) __bf16 v16bf;
typedef __attribute__((ext_vector_type(8)))  __bf16 v8bf;
typedef __attribute__((ext_vector_type(8)))  float  v8f;

#define NNODE 2048
#define NFEAT 128
#define BFEAT 16
#define HDIM  512
#define NHEAD 8
#define HEADD 64
#define NEDGE 65536
#define NDEPTH 5

static __device__ __forceinline__ unsigned pack2bf(float a, float b) {
  unsigned ua = __builtin_bit_cast(unsigned, a);
  unsigned ub = __builtin_bit_cast(unsigned, b);
  ua += 0x7FFFu + ((ua >> 16) & 1u);     // round-to-nearest-even
  ub += 0x7FFFu + ((ub >> 16) & 1u);
  return (ua >> 16) | (ub & 0xFFFF0000u);
}

// ---------------------------------------------------------------------------
// Tiled GEMM: C[M,Nc] = epi( A[M,K] @ op(B) ), op(B)=B (NN) or B^T (NT).
// 256 threads / 8 waves; 128x64 block tile; K-step 32.
// Each wave owns a 2x2 group of 16x16 WMMA subtiles:
//   4 v_wmma per k-step from 8 ds_load_b128 (A,B fragments reused 2x each).
// EPI: 0 = none, 1 = +bias[n], 2 = relu(+bias[n]), 3 = *alpha
// ---------------------------------------------------------------------------
template <bool TRANSB, int EPI>
__global__ __launch_bounds__(256) void gemm_wmma(
    const float* __restrict__ A, const float* __restrict__ B,
    const float* __restrict__ bias, float* __restrict__ C,
    int M, int Nc, int Kd, int lda, int ldb, int ldc, float alpha)
{
  constexpr int BM = 128, BN = 64, BK = 32, LP = BK + 8;  // 80B row stride
  __shared__ alignas(16) __bf16 As[BM][LP];               // As[m][k]
  __shared__ alignas(16) __bf16 Bs[BN][LP];               // Bs[n][k] (B^T tile)
  const int tid   = threadIdx.x;
  const int lane  = tid & 31;
  const int wave  = tid >> 5;
  const int bm    = blockIdx.y * BM;
  const int bn    = blockIdx.x * BN;
  const int mi0   = (wave >> 1) * 2;    // 2 of 8 M-subtiles
  const int ni0   = (wave & 1) * 2;     // 2 of 4 N-subtiles
  const int mcol  = lane & 15;
  const int khalf = lane >> 4;
  v8f acc00 = {}, acc01 = {}, acc10 = {}, acc11 = {};
  union V16 { v16bf v; v8bf h[2]; };

  for (int k0 = 0; k0 < Kd; k0 += BK) {
    { // A tile 128x32, f32 -> packed bf16, 8B LDS stores
      const int r = tid >> 3, c = (tid & 7) * 4;
      #pragma unroll
      for (int rr = 0; rr < BM; rr += 32) {
        const float4 f = *(const float4*)(A + (size_t)(bm + r + rr) * lda + (k0 + c));
        *(uint2*)&As[r + rr][c] = make_uint2(pack2bf(f.x, f.y), pack2bf(f.z, f.w));
      }
    }
    if (TRANSB) { // B is [Nc, K]; B^T tile is a direct row copy
      const int r = tid >> 3, c = (tid & 7) * 4;
      #pragma unroll
      for (int rr = 0; rr < BN; rr += 32) {
        const float4 f = *(const float4*)(B + (size_t)(bn + r + rr) * ldb + (k0 + c));
        *(uint2*)&Bs[r + rr][c] = make_uint2(pack2bf(f.x, f.y), pack2bf(f.z, f.w));
      }
    } else {
      // B is [K, Nc]; each thread loads 2 consecutive k-rows so the
      // transposed store packs (k, k+1) pairs -> single b32 per n.
      const int kk = (tid >> 4) * 2, c = (tid & 15) * 4;
      const float4 f0 = *(const float4*)(B + (size_t)(k0 + kk)     * ldb + (bn + c));
      const float4 f1 = *(const float4*)(B + (size_t)(k0 + kk + 1) * ldb + (bn + c));
      *(unsigned*)&Bs[c + 0][kk] = pack2bf(f0.x, f1.x);
      *(unsigned*)&Bs[c + 1][kk] = pack2bf(f0.y, f1.y);
      *(unsigned*)&Bs[c + 2][kk] = pack2bf(f0.z, f1.z);
      *(unsigned*)&Bs[c + 3][kk] = pack2bf(f0.w, f1.w);
    }
    __syncthreads();
    // Fragments per ISA 7.12.2 (wave32): A lane m=lane&15; elements 0-7 ->
    // k = khalf*8+0..7, elements 8-15 -> k = 16+khalf*8+0..7 (two b128 loads).
    // B lane n=lane&15; elements 0-7 -> k = khalf*8+0..7, 8-15 -> k = 16+khalf*8+0..7 (two b128 loads).
    V16 a0, a1, b0, b1;
    const __bf16* ar0 = &As[mi0 * 16 + mcol][0];
    const __bf16* ar1 = &As[(mi0 + 1) * 16 + mcol][0];
    a0.h[0] = *(const v8bf*)(ar0 + khalf * 8);
    a0.h[1] = *(const v8bf*)(ar0 + 16 + khalf * 8);
    a1.h[0] = *(const v8bf*)(ar1 + khalf * 8);
    a1.h[1] = *(const v8bf*)(ar1 + 16 + khalf * 8);
    const __bf16* br0 = &Bs[ni0 * 16 + mcol][khalf * 8];
    const __bf16* br1 = &Bs[(ni0 + 1) * 16 + mcol][khalf * 8];
    b0.h[0] = *(const v8bf*)(br0);
    b0.h[1] = *(const v8bf*)(br0 + 16);
    b1.h[0] = *(const v8bf*)(br1);
    b1.h[1] = *(const v8bf*)(br1 + 16);
    acc00 = __builtin_amdgcn_wmma_f32_16x16x32_bf16(false, a0.v, false, b0.v, (short)0, acc00, false, false);
    acc01 = __builtin_amdgcn_wmma_f32_16x16x32_bf16(false, a0.v, false, b1.v, (short)0, acc01, false, false);
    acc10 = __builtin_amdgcn_wmma_f32_16x16x32_bf16(false, a1.v, false, b0.v, (short)0, acc10, false, false);
    acc11 = __builtin_amdgcn_wmma_f32_16x16x32_bf16(false, a1.v, false, b1.v, (short)0, acc11, false, false);
    __syncthreads();
  }
  // C/D layout: VGPR r -> M = r + 8*khalf, N = lane&15
  const int c0 = bn + ni0 * 16 + mcol;
  const int c1 = c0 + 16;
  float bias0 = 0.f, bias1 = 0.f;
  if (EPI == 1 || EPI == 2) { bias0 = bias[c0]; bias1 = bias[c1]; }
  #pragma unroll
  for (int r = 0; r < 8; ++r) {
    const int row0 = bm + mi0 * 16 + r + khalf * 8;
    const int row1 = row0 + 16;
    float v00 = acc00[r], v01 = acc01[r], v10 = acc10[r], v11 = acc11[r];
    if (EPI == 3) { v00 *= alpha; v01 *= alpha; v10 *= alpha; v11 *= alpha; }
    if (EPI == 1 || EPI == 2) { v00 += bias0; v01 += bias1; v10 += bias0; v11 += bias1; }
    if (EPI == 2) {
      v00 = fmaxf(v00, 0.f); v01 = fmaxf(v01, 0.f);
      v10 = fmaxf(v10, 0.f); v11 = fmaxf(v11, 0.f);
    }
    C[(size_t)row0 * ldc + c0] = v00;
    C[(size_t)row0 * ldc + c1] = v01;
    C[(size_t)row1 * ldc + c0] = v10;
    C[(size_t)row1 * ldc + c1] = v11;
  }
}

// e_bias[E, NH] = edge_attr[E, BF] @ We[BF, NH] + be
__global__ __launch_bounds__(256) void edge_bias_kernel(
    const float* __restrict__ ea, const float* __restrict__ We,
    const float* __restrict__ be, float* __restrict__ eb)
{
  const int e = blockIdx.x * 256 + threadIdx.x;
  float a[BFEAT];
  #pragma unroll
  for (int j = 0; j < BFEAT; ++j) a[j] = ea[(size_t)e * BFEAT + j];
  #pragma unroll
  for (int h = 0; h < NHEAD; ++h) {
    float s = be[h];
    #pragma unroll
    for (int j = 0; j < BFEAT; ++j) s += a[j] * We[j * NHEAD + h];
    eb[(size_t)e * NHEAD + h] = s;
  }
}

// S[src, dst] += e_bias[e, head]  (scores already hold alpha*Q@K^T)
__global__ __launch_bounds__(256) void scatter_bias_kernel(
    const float* __restrict__ eb, const int* __restrict__ src,
    const int* __restrict__ dst, float* __restrict__ S, int head)
{
  const int e = blockIdx.x * 256 + threadIdx.x;
  atomicAdd(&S[(size_t)src[e] * NNODE + dst[e]], eb[(size_t)e * NHEAD + head]);
}

// In-place row softmax over `cols` (block = row)
__global__ __launch_bounds__(256) void softmax_rows(float* __restrict__ S, int cols)
{
  float* row = S + (size_t)blockIdx.x * cols;
  __shared__ float red[256];
  const int t = threadIdx.x;
  float m = -3.4e38f;
  for (int i = t; i < cols; i += 256) m = fmaxf(m, row[i]);
  red[t] = m; __syncthreads();
  for (int s = 128; s > 0; s >>= 1) { if (t < s) red[t] = fmaxf(red[t], red[t + s]); __syncthreads(); }
  m = red[0]; __syncthreads();
  float sum = 0.f;
  for (int i = t; i < cols; i += 256) { float e = __expf(row[i] - m); row[i] = e; sum += e; }
  red[t] = sum; __syncthreads();
  for (int s = 128; s > 0; s >>= 1) { if (t < s) red[t] += red[t + s]; __syncthreads(); }
  const float inv = 1.0f / red[0];
  for (int i = t; i < cols; i += 256) row[i] *= inv;
}

// out[row] = LayerNorm(hin[row] + delta[row]) * g + b   (H = 512, 2 elems/thread)
__global__ __launch_bounds__(256) void add_ln_kernel(
    const float* __restrict__ hin, const float* __restrict__ delta,
    const float* __restrict__ g, const float* __restrict__ b,
    float* __restrict__ outp)
{
  const int row = blockIdx.x, t = threadIdx.x;
  const size_t base = (size_t)row * HDIM;
  const float x0 = hin[base + t] + delta[base + t];
  const float x1 = hin[base + t + 256] + delta[base + t + 256];
  __shared__ float rs[256], rq[256];
  rs[t] = x0 + x1; rq[t] = x0 * x0 + x1 * x1;
  __syncthreads();
  for (int s = 128; s > 0; s >>= 1) {
    if (t < s) { rs[t] += rs[t + s]; rq[t] += rq[t + s]; }
    __syncthreads();
  }
  const float mean = rs[0] * (1.0f / HDIM);
  const float var  = rq[0] * (1.0f / HDIM) - mean * mean;
  const float rstd = rsqrtf(var + 1e-5f);
  outp[base + t]       = (x0 - mean) * rstd * g[t] + b[t];
  outp[base + t + 256] = (x1 - mean) * rstd * g[t + 256] + b[t + 256];
}

extern "C" void kernel_launch(void* const* d_in, const int* in_sizes, int n_in,
                              void* d_out, int out_size, void* d_ws, size_t ws_size,
                              hipStream_t stream)
{
  (void)in_sizes; (void)n_in; (void)out_size; (void)ws_size;
  const float* x   = (const float*)d_in[0];
  const int*   ei  = (const int*)  d_in[1];
  const float* ea  = (const float*)d_in[2];
  const float* Wn  = (const float*)d_in[3];
  const float* bn  = (const float*)d_in[4];
  const float* We  = (const float*)d_in[5];
  const float* be  = (const float*)d_in[6];
  const float* Wq  = (const float*)d_in[7];
  const float* bq  = (const float*)d_in[8];
  const float* Wk  = (const float*)d_in[9];
  const float* bk  = (const float*)d_in[10];
  const float* Wv  = (const float*)d_in[11];
  const float* bv  = (const float*)d_in[12];
  const float* Wo  = (const float*)d_in[13];
  const float* bo  = (const float*)d_in[14];
  const float* W1  = (const float*)d_in[15];
  const float* b1  = (const float*)d_in[16];
  const float* W2  = (const float*)d_in[17];
  const float* b2  = (const float*)d_in[18];
  const float* g1  = (const float*)d_in[19];
  const float* be1 = (const float*)d_in[20];
  const float* g2  = (const float*)d_in[21];
  const float* be2 = (const float*)d_in[22];
  float* out = (float*)d_out;

  float* p    = (float*)d_ws;
  float* hbuf = p; p += (size_t)NNODE * HDIM;
  float* tmp  = p; p += (size_t)NNODE * HDIM;
  float* Qb   = p; p += (size_t)NNODE * HDIM;
  float* Kb   = p; p += (size_t)NNODE * HDIM;
  float* Vb   = p; p += (size_t)NNODE * HDIM;
  float* Ob   = p; p += (size_t)NNODE * HDIM;
  float* eb   = p; p += (size_t)NEDGE * NHEAD;
  float* ff1  = p; p += (size_t)NNODE * 4 * HDIM;
  float* Sb   = p; p += (size_t)NNODE * NNODE;   // 16 MB, L2-resident

  const int* src = ei;
  const int* dst = ei + NEDGE;
  const dim3 blk(256);
  const float scale = 0.125f;   // 1/sqrt(64)

  // h = x @ Wn + bn
  gemm_wmma<false,1><<<dim3(HDIM/64, NNODE/128), blk, 0, stream>>>(
      x, Wn, bn, hbuf, NNODE, HDIM, NFEAT, NFEAT, HDIM, HDIM, 1.f);
  // e_bias = edge_attr @ We + be
  edge_bias_kernel<<<NEDGE/256, blk, 0, stream>>>(ea, We, be, eb);

  for (int layer = 0; layer < NDEPTH; ++layer) {
    gemm_wmma<false,1><<<dim3(HDIM/64, NNODE/128), blk, 0, stream>>>(
        hbuf, Wq, bq, Qb, NNODE, HDIM, HDIM, HDIM, HDIM, HDIM, 1.f);
    gemm_wmma<false,1><<<dim3(HDIM/64, NNODE/128), blk, 0, stream>>>(
        hbuf, Wk, bk, Kb, NNODE, HDIM, HDIM, HDIM, HDIM, HDIM, 1.f);
    gemm_wmma<false,1><<<dim3(HDIM/64, NNODE/128), blk, 0, stream>>>(
        hbuf, Wv, bv, Vb, NNODE, HDIM, HDIM, HDIM, HDIM, HDIM, 1.f);

    for (int hh = 0; hh < NHEAD; ++hh) {
      // S = scale * Q_h @ K_h^T     [2048 x 2048]
      gemm_wmma<true,3><<<dim3(NNODE/64, NNODE/128), blk, 0, stream>>>(
          Qb + hh * HEADD, Kb + hh * HEADD, nullptr, Sb,
          NNODE, NNODE, HEADD, HDIM, HDIM, NNODE, scale);
      // S[src,dst] += e_bias[:,hh]
      scatter_bias_kernel<<<NEDGE/256, blk, 0, stream>>>(eb, src, dst, Sb, hh);
      softmax_rows<<<NNODE, blk, 0, stream>>>(Sb, NNODE);
      // O_h = S @ V_h               [2048 x 64]
      gemm_wmma<false,0><<<dim3(HEADD/64, NNODE/128), blk, 0, stream>>>(
          Sb, Vb + hh * HEADD, nullptr, Ob + hh * HEADD,
          NNODE, HEADD, NNODE, NNODE, HDIM, HDIM, 1.f);
    }

    // tmp = O @ Wo + bo ; h = LN(h + tmp)
    gemm_wmma<false,1><<<dim3(HDIM/64, NNODE/128), blk, 0, stream>>>(
        Ob, Wo, bo, tmp, NNODE, HDIM, HDIM, HDIM, HDIM, HDIM, 1.f);
    add_ln_kernel<<<NNODE, blk, 0, stream>>>(hbuf, tmp, g1, be1, hbuf);

    // FFN
    gemm_wmma<false,2><<<dim3(4*HDIM/64, NNODE/128), blk, 0, stream>>>(
        hbuf, W1, b1, ff1, NNODE, 4*HDIM, HDIM, HDIM, 4*HDIM, 4*HDIM, 1.f);
    gemm_wmma<false,1><<<dim3(HDIM/64, NNODE/128), blk, 0, stream>>>(
        ff1, W2, b2, tmp, NNODE, HDIM, 4*HDIM, 4*HDIM, HDIM, HDIM, 1.f);
    add_ln_kernel<<<NNODE, blk, 0, stream>>>(
        hbuf, tmp, g2, be2, (layer == NDEPTH - 1) ? out : hbuf);
  }
}
